// GCNLayer_62560493633743
// MI455X (gfx1250) — compile-verified, matrix operand order checked by reading, not run on hardware
//
#include <hip/hip_runtime.h>
#include <hip/hip_bf16.h>
#include <stdint.h>

#define B_BATCH 8
#define N_NODES 2048
#define C_DIM   128

typedef __bf16 bf16_t;
typedef __attribute__((ext_vector_type(16))) bf16_t v16bf;
typedef __attribute__((ext_vector_type(8)))  bf16_t v8bf;
typedef __attribute__((ext_vector_type(8)))  float  v8f;

union FragBF { v16bf v; v8bf h[2]; };

__device__ __forceinline__ unsigned short f32_to_bf16(float f) {
    unsigned u = __float_as_uint(f);
    u += 0x7FFFu + ((u >> 16) & 1u);          // round-to-nearest-even
    return (unsigned short)(u >> 16);
}
__device__ __forceinline__ float bf16_to_f32(unsigned short h) {
    return __uint_as_float(((unsigned)h) << 16);
}

// ---------------------------------------------------------------------------
// Kernel A: per-row degree d[r] = rsqrt(1 + sum(adj[r,:])) and adj -> bf16
// ---------------------------------------------------------------------------
__global__ void gcn_prep(const float* __restrict__ adj,
                         float* __restrict__ deg,
                         unsigned short* __restrict__ adjb) {
    __shared__ float red[256];
    const int r = blockIdx.x;
    const int t = threadIdx.x;
    const float4* rowp = (const float4*)(adj + (size_t)r * N_NODES);
    float4 a = rowp[t * 2 + 0];
    float4 b = rowp[t * 2 + 1];
    float s = a.x + a.y + a.z + a.w + b.x + b.y + b.z + b.w;

    union { unsigned short u16[8]; uint4 q; } pk;
    pk.u16[0] = f32_to_bf16(a.x); pk.u16[1] = f32_to_bf16(a.y);
    pk.u16[2] = f32_to_bf16(a.z); pk.u16[3] = f32_to_bf16(a.w);
    pk.u16[4] = f32_to_bf16(b.x); pk.u16[5] = f32_to_bf16(b.y);
    pk.u16[6] = f32_to_bf16(b.z); pk.u16[7] = f32_to_bf16(b.w);
    *(uint4*)(adjb + (size_t)r * N_NODES + t * 8) = pk.q;

    red[t] = s;
    __syncthreads();
    for (int off = 128; off > 0; off >>= 1) {
        if (t < off) red[t] += red[t + off];
        __syncthreads();
    }
    if (t == 0) deg[r] = rsqrtf(red[0] + 1.0f);
}

// ---------------------------------------------------------------------------
// Kernel B: y_t[b][c][m] = bf16( d[m] * sum_k x[b,m,k] * W[c,k] )
// GEMM (B*N=16384) x 128 x 128 via bf16 WMMA, f32 accumulate.
// ---------------------------------------------------------------------------
__global__ void gcn_proj(const float* __restrict__ x,    // (16384, 128)
                         const float* __restrict__ W,    // (128, 128)
                         const float* __restrict__ deg,  // (2048)
                         unsigned short* __restrict__ yt) // (8, 128, 2048)
{
    __shared__ __attribute__((aligned(16))) unsigned short xs[64 * C_DIM];
    __shared__ __attribute__((aligned(16))) unsigned short wt[C_DIM * C_DIM];

    const int tid  = threadIdx.x;
    const int lane = tid & 31;
    const int wave = tid >> 5;
    const int m0   = blockIdx.x * 64;          // row in flattened (B*N)

    // load x tile 64x128 (f32 -> bf16)
    for (int it = 0; it < 8; ++it) {
        int idx = tid + it * 256;              // 2048 float4 chunks
        int rr = idx >> 5;
        int cc = (idx & 31) * 4;
        float4 v = *(const float4*)(x + (size_t)(m0 + rr) * C_DIM + cc);
        union { unsigned short u[4]; uint2 q; } pk;
        pk.u[0] = f32_to_bf16(v.x); pk.u[1] = f32_to_bf16(v.y);
        pk.u[2] = f32_to_bf16(v.z); pk.u[3] = f32_to_bf16(v.w);
        *(uint2*)&xs[rr * C_DIM + cc] = pk.q;
    }
    // load full W 128x128 (f32 -> bf16)
    for (int it = 0; it < 16; ++it) {
        int idx = tid + it * 256;              // 4096 float4 chunks
        int rr = idx >> 5;
        int cc = (idx & 31) * 4;
        float4 v = *(const float4*)(W + (size_t)rr * C_DIM + cc);
        union { unsigned short u[4]; uint2 q; } pk;
        pk.u[0] = f32_to_bf16(v.x); pk.u[1] = f32_to_bf16(v.y);
        pk.u[2] = f32_to_bf16(v.z); pk.u[3] = f32_to_bf16(v.w);
        *(uint2*)&wt[rr * C_DIM + cc] = pk.q;
    }
    __syncthreads();

    const int Mbase = (wave >> 1) * 16;
    const int Nbase = (wave & 1) * 64;
    const int half  = lane >> 4;
    const int l16   = lane & 15;

    v8f zero = {0.f, 0.f, 0.f, 0.f, 0.f, 0.f, 0.f, 0.f};
    v8f acc[4] = {zero, zero, zero, zero};

    for (int ks = 0; ks < C_DIM; ks += 32) {
        FragBF A;
        const unsigned short* ap = &xs[(Mbase + l16) * C_DIM + ks + half * 8];
        A.h[0] = *(const v8bf*)(ap);
        A.h[1] = *(const v8bf*)(ap + 16);
        for (int j = 0; j < 4; ++j) {
            FragBF Bf;
            const unsigned short* bp =
                &wt[(Nbase + j * 16 + l16) * C_DIM + ks + half * 8];
            Bf.h[0] = *(const v8bf*)(bp);
            Bf.h[1] = *(const v8bf*)(bp + 16);
            acc[j] = __builtin_amdgcn_wmma_f32_16x16x32_bf16(
                false, A.v, false, Bf.v, (short)0, acc[j], false, false);
        }
    }

    // epilogue: scale by d[m], store transposed as bf16
    for (int j = 0; j < 4; ++j) {
        int c = Nbase + j * 16 + l16;
        for (int i = 0; i < 8; ++i) {
            int mt = m0 + Mbase + half * 8 + i;   // 0..16383
            int b  = mt >> 11;
            int m  = mt & (N_NODES - 1);
            float val = acc[j][i] * deg[m];
            yt[((size_t)b * C_DIM + c) * N_NODES + m] = f32_to_bf16(val);
        }
    }
}

// ---------------------------------------------------------------------------
// Kernel C: out[b,n,c] = d[n] * (sum_m adj[n,m] * y[b,m,c] + y[b,n,c]) + bias[c]
// Per-batch GEMM 2048 x 128 x 2048 via bf16 WMMA.
// Double-buffered LDS pipeline fed by GLOBAL_LOAD_ASYNC_TO_LDS_B128 (ASYNCcnt).
// ---------------------------------------------------------------------------
__global__ void gcn_agg(const unsigned short* __restrict__ adjb, // (2048,2048)
                        const unsigned short* __restrict__ yt,   // (8,128,2048)
                        const float* __restrict__ deg,
                        const float* __restrict__ bias,
                        float* __restrict__ out)                  // (8,2048,128)
{
    __shared__ __attribute__((aligned(16))) unsigned short at[2][64 * 64];
    __shared__ __attribute__((aligned(16))) unsigned short bt[2][C_DIM * 64];

    const int tid  = threadIdx.x;
    const int lane = tid & 31;
    const int wave = tid >> 5;
    const int n0   = blockIdx.x * 64;
    const int b    = blockIdx.y;

    const int Mbase = (wave >> 1) * 16;
    const int Nbase = (wave & 1) * 64;
    const int half  = lane >> 4;
    const int l16   = lane & 15;

    const unsigned short* ybase = yt + (size_t)b * C_DIM * N_NODES;

    // Per-thread tile-chunk coordinates (16B chunks).
    // adj tile 64x64: 512 chunks -> 2 per thread; y tile 128x64: 1024 -> 4.
    int a_rr[2], a_cc[2], y_rr[4], y_cc[4];
    for (int it = 0; it < 2; ++it) {
        int idx = tid + it * 256;
        a_rr[it] = idx >> 3;
        a_cc[it] = (idx & 7) * 8;
    }
    for (int it = 0; it < 4; ++it) {
        int idx = tid + it * 256;
        y_rr[it] = idx >> 3;
        y_cc[it] = (idx & 7) * 8;
    }

    // Issue async global->LDS copies for tile at k0 into buffer `buf`.
    auto issue_tile = [&](int buf, int k0) {
        for (int it = 0; it < 2; ++it) {
            unsigned lds  = (unsigned)(uintptr_t)&at[buf][a_rr[it] * 64 + a_cc[it]];
            unsigned goff = (unsigned)((((n0 + a_rr[it]) * N_NODES) + k0 + a_cc[it]) * 2);
            asm volatile("global_load_async_to_lds_b128 %0, %1, %2"
                         :: "v"(lds), "v"(goff), "s"(adjb) : "memory");
        }
        for (int it = 0; it < 4; ++it) {
            unsigned lds  = (unsigned)(uintptr_t)&bt[buf][y_rr[it] * 64 + y_cc[it]];
            unsigned goff = (unsigned)(((y_rr[it] * N_NODES) + k0 + y_cc[it]) * 2);
            asm volatile("global_load_async_to_lds_b128 %0, %1, %2"
                         :: "v"(lds), "v"(goff), "s"(ybase) : "memory");
        }
    };

    v8f zero = {0.f, 0.f, 0.f, 0.f, 0.f, 0.f, 0.f, 0.f};
    v8f acc[4] = {zero, zero, zero, zero};

    // Prologue: fill buffer 0.
    issue_tile(0, 0);
    asm volatile("s_wait_asynccnt 0" ::: "memory");
    __syncthreads();

    int buf = 0;
    for (int k0 = 0; k0 < N_NODES; k0 += 64) {
        // Prefetch next tile into the other buffer while computing this one.
        if (k0 + 64 < N_NODES) issue_tile(buf ^ 1, k0 + 64);

        for (int ks = 0; ks < 64; ks += 32) {
            FragBF A;
            const unsigned short* ap = &at[buf][(Mbase + l16) * 64 + ks + half * 8];
            A.h[0] = *(const v8bf*)(ap);
            A.h[1] = *(const v8bf*)(ap + 16);
            for (int j = 0; j < 4; ++j) {
                FragBF Bf;
                const unsigned short* bp =
                    &bt[buf][(Nbase + j * 16 + l16) * 64 + ks + half * 8];
                Bf.h[0] = *(const v8bf*)(bp);
                Bf.h[1] = *(const v8bf*)(bp + 16);
                acc[j] = __builtin_amdgcn_wmma_f32_16x16x32_bf16(
                    false, A.v, false, Bf.v, (short)0, acc[j], false, false);
            }
        }

        // Wave-local async completion, then block-wide visibility + compute fence.
        asm volatile("s_wait_asynccnt 0" ::: "memory");
        __syncthreads();
        buf ^= 1;
    }

    // epilogue: identity term, d[n] scaling, bias, fp32 store
    for (int j = 0; j < 4; ++j) {
        int c  = Nbase + j * 16 + l16;
        float bs = bias[c];
        for (int i = 0; i < 8; ++i) {
            int n = n0 + Mbase + half * 8 + i;
            float ident = bf16_to_f32(ybase[(size_t)c * N_NODES + n]);
            float val = deg[n] * (acc[j][i] + ident) + bs;
            out[((size_t)b * N_NODES + n) * C_DIM + c] = val;
        }
    }
}

// ---------------------------------------------------------------------------
extern "C" void kernel_launch(void* const* d_in, const int* in_sizes, int n_in,
                              void* d_out, int out_size, void* d_ws, size_t ws_size,
                              hipStream_t stream) {
    const float* x    = (const float*)d_in[0];   // (8, 2048, 128)
    const float* adj  = (const float*)d_in[1];   // (2048, 2048)
    const float* W    = (const float*)d_in[2];   // (128, 128)
    const float* bias = (const float*)d_in[3];   // (128)
    float* out = (float*)d_out;

    char* ws = (char*)d_ws;
    float* deg           = (float*)ws;                                   // 8 KB
    unsigned short* adjb = (unsigned short*)(ws + 8192);                 // 8 MB
    unsigned short* yt   = (unsigned short*)(ws + 8192 +
                               (size_t)N_NODES * N_NODES * sizeof(unsigned short)); // 4 MB

    gcn_prep<<<N_NODES, 256, 0, stream>>>(adj, deg, adjb);
    gcn_proj<<<(B_BATCH * N_NODES) / 64, 256, 0, stream>>>(x, W, deg, yt);
    dim3 gridC(N_NODES / 64, B_BATCH);
    gcn_agg<<<gridC, 256, 0, stream>>>(adjb, yt, deg, bias, out);
}
